// HybridTokenPruner_66262755442837
// MI455X (gfx1250) — matrix operand order repaired and checked by reading, not compiled, not run on hardware
//
#include <hip/hip_runtime.h>

#define HIDDEN   1024
#define HID4     256
#define KEEP     144
#define NCOMP    36
#define BATCH    64
#define SEQ      576
#define RJ       (SEQ - KEEP)      // 432
#define GRP      (RJ / NCOMP)      // 12
#define OUTROWS  (KEEP + NCOMP)    // 180

#define TOK_PER_BLK 64
#define LDS_STRIDE  (HIDDEN + 8)   // bf16 elements; 2064B row stride = 4-bank skew

typedef __attribute__((ext_vector_type(16))) __bf16 v16bf;
typedef __attribute__((ext_vector_type(8)))  float  v8f;
typedef __attribute__((ext_vector_type(8)))  unsigned int v8u;

union Frag { v16bf bf; v8u u; };

// Pack two f32 into one dword of two bf16 (round-to-nearest) using v_perm_b32.
__device__ inline unsigned pack_bf16(float lo, float hi) {
    unsigned a = __builtin_bit_cast(unsigned, lo) + 0x8000u;
    unsigned b = __builtin_bit_cast(unsigned, hi) + 0x8000u;
    return __builtin_amdgcn_perm(b, a, 0x07060302u);   // (bf16(hi)<<16)|bf16(lo)
}

// ---------------------------------------------------------------------------
// k0: W1 (1024 x 256, f32, [k][f]) -> W1t (256 x 1024, bf16, [f][k]) so WMMA
// B-fragments are contiguous 32B per lane.  512 KB -> stays L2-resident.
// ---------------------------------------------------------------------------
__global__ void __launch_bounds__(256) k0_w1t(const float* __restrict__ W1,
                                              unsigned short* __restrict__ W1t) {
    int idx = blockIdx.x * 256 + threadIdx.x;     // 0 .. 256*1024-1
    int f = idx >> 10;
    int k = idx & 1023;
    unsigned u = __builtin_bit_cast(unsigned, W1[k * HID4 + f]) + 0x8000u;
    W1t[f * HIDDEN + k] = (unsigned short)(u >> 16);
}

// ---------------------------------------------------------------------------
// k1: fused score GEMM.  Block = 128 threads (4 waves) = 64 tokens.
// Phase 1: cooperatively convert the 64x1024 f32 x tile to bf16 in LDS (once).
// Phase 2: each wave owns 16 tokens; 16 feature tiles x 32 K-steps of
//          v_wmma_f32_16x16x32_bf16, A from LDS (2x ds_load_b128), B from
//          L2-resident W1t (2x global_load_b128).
//   h = relu(x@W1 + b1); logit = h@W2   (sigmoid/b2 dropped: rank-invariant)
// A-fragment (16-bit A 16x32, ISA 7.12.2): lane = row M=lane%16,
//   halves j = K(k0+16g+j)  (contiguous, matching B), g=lane/16.
// B-fragment: lane = col N=lane%16, halves j = K(k0+16g+j)  (contiguous).
// C/D: VGPR r, lane L -> token M = r + 8*(L/16), feature N = L%16.
// ---------------------------------------------------------------------------
__global__ void __launch_bounds__(128) k1_scores(const float* __restrict__ x,
                                                 const unsigned short* __restrict__ W1t,
                                                 const float* __restrict__ b1,
                                                 const float* __restrict__ W2,
                                                 float* __restrict__ scores) {
    __shared__ __align__(16) unsigned short xs[TOK_PER_BLK * LDS_STRIDE]; // 132 KB

    const int tid  = threadIdx.x;
    const int tok0 = blockIdx.x * TOK_PER_BLK;

    // ---- Phase 1: f32 -> bf16 conversion into LDS, fully coalesced --------
    // 64 rows x 128 chunks of 8 floats; consecutive tids take consecutive
    // chunks within a row (128B-coalesced), one ds_store_b128 per chunk.
    for (int it = 0; it < (TOK_PER_BLK * HIDDEN / 8) / 128; ++it) {
        int c   = tid + 128 * it;
        int row = c >> 7;
        int col = (c & 127) << 3;
        const float4* p =
            reinterpret_cast<const float4*>(x + (size_t)(tok0 + row) * HIDDEN + col);
        float4 a = p[0], b = p[1];
        uint4 d;
        d.x = pack_bf16(a.x, a.y); d.y = pack_bf16(a.z, a.w);
        d.z = pack_bf16(b.x, b.y); d.w = pack_bf16(b.z, b.w);
        *reinterpret_cast<uint4*>(&xs[row * LDS_STRIDE + col]) = d;
    }
    __syncthreads();

    // ---- Phase 2: WMMA over LDS-resident A ---------------------------------
    const int lane  = tid & 31;
    const int wave  = tid >> 5;
    const int group = lane >> 4;
    const int nl    = lane & 15;
    const int m0    = tok0 + wave * 16;

    const unsigned short* arow = &xs[(wave * 16 + nl) * LDS_STRIDE];

    float acc[8];
#pragma unroll
    for (int r = 0; r < 8; ++r) acc[r] = 0.f;

    for (int t = 0; t < 16; ++t) {                    // feature tiles (N)
        const int f0 = t * 16;
        v8f c = {};
        const unsigned short* wrow = W1t + (size_t)(f0 + nl) * HIDDEN + group * 16;
        if (t < 15)  // warm L0/L2 for the next feature tile (global_prefetch_b8)
            __builtin_prefetch(W1t + (size_t)(f0 + 16 + nl) * HIDDEN, 0, 1);
#pragma unroll 4
        for (int k0 = 0; k0 < HIDDEN; k0 += 32) {     // K loop
            Frag A, B;
            uint4 lo = *reinterpret_cast<const uint4*>(arow + k0 + 16 * group);
            uint4 hi = *reinterpret_cast<const uint4*>(arow + k0 + 16 * group + 8);
            A.u[0] = lo.x; A.u[1] = lo.y; A.u[2] = lo.z; A.u[3] = lo.w;
            A.u[4] = hi.x; A.u[5] = hi.y; A.u[6] = hi.z; A.u[7] = hi.w;
            B.u = *reinterpret_cast<const v8u*>(wrow + k0);
            c = __builtin_amdgcn_wmma_f32_16x16x32_bf16(false, A.bf, false, B.bf,
                                                        (short)0, c, false, false);
        }
        const float bb = b1[f0 + nl];
        const float w2 = W2[f0 + nl];
#pragma unroll
        for (int r = 0; r < 8; ++r) {
            float h = c[r] + bb;
            acc[r] += fmaxf(h, 0.f) * w2;
        }
    }
    // Reduce across the 16 feature lanes within each half-wave (masks < 16).
#pragma unroll
    for (int r = 0; r < 8; ++r) {
        float v = acc[r];
        v += __shfl_xor(v, 1, 32);
        v += __shfl_xor(v, 2, 32);
        v += __shfl_xor(v, 4, 32);
        v += __shfl_xor(v, 8, 32);
        acc[r] = v;
    }
    if (nl == 0) {
#pragma unroll
        for (int r = 0; r < 8; ++r)
            scores[m0 + group * 8 + r] = acc[r];      // token = m0 + 8*group + r
    }
}

// ---------------------------------------------------------------------------
// k2: exact top-K per batch row via rank counting (tie-break: lower index wins,
// matching jax.lax.top_k), then stable prefix sums produce kept/rejected
// indices in original token order (matching sorted topk_idx + stable argsort).
// ---------------------------------------------------------------------------
__global__ void __launch_bounds__(SEQ) k2_topk(const float* __restrict__ scores,
                                               int* __restrict__ row_map,
                                               int* __restrict__ rej_map) {
    __shared__ float s[SEQ];
    __shared__ int   kf[SEQ];
    const int b = blockIdx.x;
    const int i = threadIdx.x;
    const float si = scores[b * SEQ + i];
    s[i] = si;
    __syncthreads();
    int rank = 0;
    for (int j = 0; j < SEQ; ++j) {
        float sj = s[j];
        rank += (sj > si) || (sj == si && j < i);
    }
    const int keep = (rank < KEEP) ? 1 : 0;
    kf[i] = keep;
    __syncthreads();
    int before = 0;
    for (int j = 0; j < i; ++j) before += kf[j];
    if (keep) row_map[b * KEEP + before]       = i;
    else      rej_map[b * RJ   + (i - before)] = i;
}

// ---------------------------------------------------------------------------
// k3: one block per output row. Kept rows: float4 copy of original f32 x.
// Compressed rows: mean of 12 rejected rows.
// ---------------------------------------------------------------------------
__global__ void __launch_bounds__(256) k3_gather(const float* __restrict__ x,
                                                 const int* __restrict__ row_map,
                                                 const int* __restrict__ rej_map,
                                                 float* __restrict__ out) {
    const int row = blockIdx.x;
    const int b   = row / OUTROWS;
    const int r   = row % OUTROWS;
    const int t   = threadIdx.x;            // 256 threads * float4 = 1024 floats
    const float* xb = x + (size_t)b * SEQ * HIDDEN;
    float4 v;
    if (r < KEEP) {
        int src = row_map[b * KEEP + r];
        v = reinterpret_cast<const float4*>(xb + (size_t)src * HIDDEN)[t];
    } else {
        int g = r - KEEP;
        float4 a; a.x = 0.f; a.y = 0.f; a.z = 0.f; a.w = 0.f;
        for (int j = 0; j < GRP; ++j) {
            int src = rej_map[b * RJ + g * GRP + j];
            float4 w = reinterpret_cast<const float4*>(xb + (size_t)src * HIDDEN)[t];
            a.x += w.x; a.y += w.y; a.z += w.z; a.w += w.w;
        }
        const float inv = 1.0f / (float)GRP;
        v.x = a.x * inv; v.y = a.y * inv; v.z = a.z * inv; v.w = a.w * inv;
    }
    reinterpret_cast<float4*>(out + (size_t)row * HIDDEN)[t] = v;
}

// ---------------------------------------------------------------------------
extern "C" void kernel_launch(void* const* d_in, const int* in_sizes, int n_in,
                              void* d_out, int out_size, void* d_ws, size_t ws_size,
                              hipStream_t stream) {
    const float* x  = (const float*)d_in[0];
    const float* W1 = (const float*)d_in[1];
    const float* b1 = (const float*)d_in[2];
    const float* W2 = (const float*)d_in[3];
    // d_in[4] = b2: unused (constant shift inside a monotone sigmoid -> rank-invariant)

    char* ws = (char*)d_ws;
    unsigned short* W1t = (unsigned short*)ws;                         // 512 KB
    float* scores  = (float*)(ws + 512 * 1024);                        // 144 KB
    int*   row_map = (int*)  (ws + 512 * 1024 + 160 * 1024);           // 64*144 ints
    int*   rej_map = row_map + BATCH * KEEP;                           // 64*432 ints
    float* out = (float*)d_out;

    k0_w1t   <<<dim3((HID4 * HIDDEN) / 256), dim3(256), 0, stream>>>(W1, W1t);
    k1_scores<<<dim3((BATCH * SEQ) / TOK_PER_BLK), dim3(128), 0, stream>>>(x, W1t, b1, W2, scores);
    k2_topk  <<<dim3(BATCH),                 dim3(SEQ), 0, stream>>>(scores, row_map, rej_map);
    k3_gather<<<dim3(BATCH * OUTROWS),       dim3(256), 0, stream>>>(x, row_map, rej_map, out);
}
